// TTS_48670569398709
// MI455X (gfx1250) — matrix operand order repaired and checked by reading, not compiled, not run on hardware
//
#include <hip/hip_runtime.h>
#include <hip/hip_bf16.h>

// ---------------------------------------------------------------------------
// MI455X (gfx1250) implementation of the Tacotron-style TTS reference.
// All dense math goes through V_WMMA_F32_16X16X32_F16 (f16 inputs, f32 acc).
// Weights pre-packed per call into the WMMA B-fragment layout (f16, chunk-
// major). GEMM/conv waves compute 2 adjacent N-tiles; all loop variants are
// template-specialized so inner loops carry no uniform branches.
// ---------------------------------------------------------------------------

typedef __attribute__((ext_vector_type(16))) _Float16 v16h;
typedef __attribute__((ext_vector_type(8)))  _Float16 v8h;
typedef __attribute__((ext_vector_type(8)))  float    v8f;
typedef __attribute__((ext_vector_type(4)))  float    v4f;

#define WMMA16(acc, a, b) \
  __builtin_amdgcn_wmma_f32_16x16x32_f16(false, (a), false, (b), (short)0, (acc), false, false)

#define ACT_NONE 0
#define ACT_RELU 1
#define ACT_SIG  2
#define ACT_TANH 3

__device__ __forceinline__ float act_f(float x, int act) {
  if (act == ACT_RELU) return x > 0.f ? x : 0.f;
  if (act == ACT_SIG)  return 1.f / (1.f + expf(-x));
  if (act == ACT_TANH) return tanhf(x);
  return x;
}

__device__ __forceinline__ v4f max4(v4f a, v4f b) {
  v4f r;
  r[0] = fmaxf(a[0], b[0]); r[1] = fmaxf(a[1], b[1]);
  r[2] = fmaxf(a[2], b[2]); r[3] = fmaxf(a[3], b[3]);
  return r;
}

// Pack 4 runs of 4 f32 into the v16h A-fragment slots [0..15].
__device__ __forceinline__ void pack16(v16h& a, v4f x0, v4f x1, v4f x2, v4f x3) {
#pragma unroll
  for (int j = 0; j < 4; ++j) {
    a[j]      = (_Float16)x0[j];
    a[4 + j]  = (_Float16)x1[j];
    a[8 + j]  = (_Float16)x2[j];
    a[12 + j] = (_Float16)x3[j];
  }
}

// B-fragment from packed f16 weights: halfs [0..7] and [16..23] (two b128 loads).
__device__ __forceinline__ v16h load_ph(const _Float16* __restrict__ p) {
  v8h b0 = *(const v8h*)p;
  v8h b1 = *(const v8h*)(p + 16);
  v16h b;
#pragma unroll
  for (int j = 0; j < 8; ++j) { b[j] = b0[j]; b[8 + j] = b1[j]; }
  return b;
}

// A-fragment, no bounds checks (K % 32 == 0 path).
__device__ __forceinline__ v16h load_a_fast(const float* __restrict__ arow, int k0, int kh) {
  v4f x0 = *(const v4f*)(arow + k0 + kh);
  v4f x1 = *(const v4f*)(arow + k0 + kh + 4);
  v4f x2 = *(const v4f*)(arow + k0 + 16 + kh);
  v4f x3 = *(const v4f*)(arow + k0 + 16 + kh + 4);
  v16h a; pack16(a, x0, x1, x2, x3);
  return a;
}

// A-fragment with per-run predicates (K % 32 != 0 tail; K multiple of 16).
__device__ __forceinline__ v16h load_a_chk(const float* __restrict__ arow,
                                           int k0, int kh, int K) {
  int r0 = k0 + kh, r1 = k0 + 16 + kh;
  v4f x0 = {}, x1 = {}, x2 = {}, x3 = {};
  if (r0 < K) { x0 = *(const v4f*)(arow + r0); x1 = *(const v4f*)(arow + r0 + 4); }
  if (r1 < K) { x2 = *(const v4f*)(arow + r1); x3 = *(const v4f*)(arow + r1 + 4); }
  v16h a; pack16(a, x0, x1, x2, x3);
  return a;
}

// Branch-free GEMM k-loop, specialized on (checked A, dual tile).
template <bool CHK, bool HAS2>
__device__ __forceinline__ void gemm_kloop(const float* __restrict__ arow,
                                           const _Float16* __restrict__ pb, long pstep,
                                           int K, int kh, v8f& acc0, v8f& acc1) {
  for (int k0 = 0; k0 < K; k0 += 32, pb += pstep) {
    v16h a;
    if (CHK) {
      a = load_a_chk(arow, k0, kh, K);
    } else {
      __builtin_prefetch(pb + pstep, 0, 0);
      a = load_a_fast(arow, k0, kh);
    }
    acc0 = WMMA16(acc0, a, load_ph(pb));
    if (HAS2) acc1 = WMMA16(acc1, a, load_ph(pb + 512));
  }
}

// ---------------------------------------------------------------------------
// Weight packer: W[width][K][N] (f32, row-major) -> P[width][nch][Npad][32] f16
// P[((tap*nch + c)*Npad + n)*32 + kk] = W[tap][c*32+kk][n], zero padded.
// ---------------------------------------------------------------------------
__global__ void pack_w_kernel(const float* __restrict__ W, _Float16* __restrict__ P,
                              int width, int K, int N, int Npad, int nch) {
  long per = (long)nch * Npad * 32;
  long idx = (long)blockIdx.x * blockDim.x + threadIdx.x;
  if (idx >= per * width) return;
  int tap = (int)(idx / per);
  long rem = idx - (long)tap * per;
  int kk = (int)(rem & 31);
  long rest = rem >> 5;
  int n = (int)(rest % Npad);
  int c = (int)(rest / Npad);
  int k = c * 32 + kk;
  float v = (k < K && n < N) ? W[((long)tap * K + k) * N + n] : 0.f;
  P[idx] = (_Float16)v;
}

// ---------------------------------------------------------------------------
// Generic GEMM, 2 N-tiles per wave: C[M,N] = act(A[M,K] @ W + bias (+resid)).
// W pre-packed (Npad cols). M multiple of 16; K multiple of 16.
// ---------------------------------------------------------------------------
__global__ void gemm_kernel(const float* __restrict__ A, const _Float16* __restrict__ P,
                            const float* __restrict__ bias, const float* __restrict__ resid,
                            float* __restrict__ C, int N, int Npad, int K, int ldc, int act) {
  const int lane = threadIdx.x;
  const int n0 = blockIdx.x * 32;
  const int m0 = blockIdx.y * 16;
  const int col = n0 + (lane & 15);
  const bool has2 = (n0 + 16) < Npad;
  const int kh = (lane >> 4) * 8;
  const int kb = (lane >> 4) * 8;
  const float* arow = A + (long)(m0 + (lane & 15)) * K;
  const _Float16* pb = P + (long)col * 32 + kb;
  const long pstep = (long)Npad * 32;
  v8f acc0 = {}, acc1 = {};
  if ((K & 31) == 0) {
    if (has2) gemm_kloop<false, true >(arow, pb, pstep, K, kh, acc0, acc1);
    else      gemm_kloop<false, false>(arow, pb, pstep, K, kh, acc0, acc1);
  } else {
    if (has2) gemm_kloop<true,  true >(arow, pb, pstep, K, kh, acc0, acc1);
    else      gemm_kloop<true,  false>(arow, pb, pstep, K, kh, acc0, acc1);
  }
  const int rb = (lane >> 4) * 8;
  if (col < N) {
    float bv = bias ? bias[col] : 0.f;
#pragma unroll
    for (int r = 0; r < 8; ++r) {
      long row = m0 + rb + r;
      float v = acc0[r] + bv;
      if (resid) v += resid[row * ldc + col];
      C[row * ldc + col] = act_f(v, act);
    }
  }
  int col2 = col + 16;
  if (has2 && col2 < N) {
    float bv = bias ? bias[col2] : 0.f;
#pragma unroll
    for (int r = 0; r < 8; ++r) {
      long row = m0 + rb + r;
      float v = acc1[r] + bv;
      if (resid) v += resid[row * ldc + col2];
      C[row * ldc + col2] = act_f(v, act);
    }
  }
}

// ---------------------------------------------------------------------------
// Conv1d 'SAME' tap+k loop, specialized on (boundary, pool, dual tile).
// BOUND=false requires: every tap row in range and Cin % 32 == 0.
// ---------------------------------------------------------------------------
template <bool BOUND, bool POOL, bool HAS2>
__device__ __forceinline__ void conv_body(const float* __restrict__ X, long rowbase,
                                          int myt, int pad_lo, int width, int Tlen, int Cin,
                                          const _Float16* __restrict__ P, int nch, int Cout,
                                          int nb, int kh, int kb, v8f& acc0, v8f& acc1) {
  const long pstep = (long)Cout * 32;
  for (int tap = 0; tap < width; ++tap) {
    int tt = myt + tap - pad_lo;
    const float* xrow  = X + (rowbase + tt) * (long)Cin;
    const float* xrow2 = xrow + Cin;
    const _Float16* pb = P + ((long)tap * nch * Cout + nb) * 32 + kb;
    bool ok = true, okp = POOL;
    if (BOUND) {
      ok  = (tt >= 0) && (tt < Tlen);
      okp = POOL && ok && (tt + 1 < Tlen);
    }
    for (int k0 = 0; k0 < Cin; k0 += 32, pb += pstep) {
      int r0 = k0 + kh, r1 = k0 + 16 + kh;
      v4f x0 = {}, x1 = {}, x2 = {}, x3 = {};
      bool in0 = !BOUND || (ok && r0 < Cin);
      bool in1 = !BOUND || (ok && r1 < Cin);
      if (in0) {
        x0 = *(const v4f*)(xrow + r0); x1 = *(const v4f*)(xrow + r0 + 4);
        if (okp) {
          x0 = max4(x0, *(const v4f*)(xrow2 + r0));
          x1 = max4(x1, *(const v4f*)(xrow2 + r0 + 4));
        }
      }
      if (in1) {
        x2 = *(const v4f*)(xrow + r1); x3 = *(const v4f*)(xrow + r1 + 4);
        if (okp) {
          x2 = max4(x2, *(const v4f*)(xrow2 + r1));
          x3 = max4(x3, *(const v4f*)(xrow2 + r1 + 4));
        }
      }
      v16h a; pack16(a, x0, x1, x2, x3);
      acc0 = WMMA16(acc0, a, load_ph(pb));
      if (HAS2) acc1 = WMMA16(acc1, a, load_ph(pb + 512));
    }
  }
}

// ---------------------------------------------------------------------------
// Conv1d 'SAME' as tap-summed GEMM, 2 N-tiles per wave. X: [32, Tlen, Cin].
// Packed weights P[width][nch][Cout][32]. Optional fused maxpool2 on input.
// Cout, Cin multiples of 16; Tlen multiple of 16.
// ---------------------------------------------------------------------------
__global__ void conv_gemm_kernel(const float* __restrict__ X, const _Float16* __restrict__ P,
                                 const float* __restrict__ bias, const float* __restrict__ resid,
                                 float* __restrict__ Y,
                                 int Tlen, int Cin, int Cout, int width, int pad_lo,
                                 int ldy, int ycol0, int act, int pool) {
  const int lane = threadIdx.x;
  const int n0 = blockIdx.x * 32;
  const int row0 = blockIdx.y * 16;      // global row = b*Tlen + t
  const int b = row0 / Tlen;
  const int t0 = row0 - b * Tlen;
  const int myt = t0 + (lane & 15);
  const int kh = (lane >> 4) * 8;
  const int kb = (lane >> 4) * 8;
  const int nb = n0 + (lane & 15);
  const bool has2 = (n0 + 16) < Cout;
  const int nch = (Cin + 31) / 32;
  const long rowbase = (long)b * Tlen;
  const bool interior =
      (t0 - pad_lo >= 0) && (t0 + 15 + (width - 1) - pad_lo + (pool ? 1 : 0)) < Tlen;
  const bool fast = interior && (Cin & 31) == 0;
  v8f acc0 = {}, acc1 = {};
  if (fast) {
    if (pool) {
      if (has2) conv_body<false, true,  true >(X, rowbase, myt, pad_lo, width, Tlen, Cin, P, nch, Cout, nb, kh, kb, acc0, acc1);
      else      conv_body<false, true,  false>(X, rowbase, myt, pad_lo, width, Tlen, Cin, P, nch, Cout, nb, kh, kb, acc0, acc1);
    } else {
      if (has2) conv_body<false, false, true >(X, rowbase, myt, pad_lo, width, Tlen, Cin, P, nch, Cout, nb, kh, kb, acc0, acc1);
      else      conv_body<false, false, false>(X, rowbase, myt, pad_lo, width, Tlen, Cin, P, nch, Cout, nb, kh, kb, acc0, acc1);
    }
  } else {
    if (pool) {
      if (has2) conv_body<true,  true,  true >(X, rowbase, myt, pad_lo, width, Tlen, Cin, P, nch, Cout, nb, kh, kb, acc0, acc1);
      else      conv_body<true,  true,  false>(X, rowbase, myt, pad_lo, width, Tlen, Cin, P, nch, Cout, nb, kh, kb, acc0, acc1);
    } else {
      if (has2) conv_body<true,  false, true >(X, rowbase, myt, pad_lo, width, Tlen, Cin, P, nch, Cout, nb, kh, kb, acc0, acc1);
      else      conv_body<true,  false, false>(X, rowbase, myt, pad_lo, width, Tlen, Cin, P, nch, Cout, nb, kh, kb, acc0, acc1);
    }
  }
  const int rb = (lane >> 4) * 8;
  {
    float bv = bias ? bias[nb] : 0.f;
    int ycol = ycol0 + nb;
#pragma unroll
    for (int r = 0; r < 8; ++r) {
      long row = row0 + rb + r;
      float v = acc0[r] + bv;
      if (resid) v += resid[row * ldy + ycol];
      Y[row * ldy + ycol] = act_f(v, act);
    }
  }
  if (has2) {
    int nb2 = nb + 16;
    float bv = bias ? bias[nb2] : 0.f;
    int ycol = ycol0 + nb2;
#pragma unroll
    for (int r = 0; r < 8; ++r) {
      long row = row0 + rb + r;
      float v = acc1[r] + bv;
      if (resid) v += resid[row * ldy + ycol];
      Y[row * ldy + ycol] = act_f(v, act);
    }
  }
}

// ---------------------------------------------------------------------------
// Persistent GRU sequence scan (dx = dh = 128, B = 32).
// One workgroup = 32 waves. Gate GEMM (32x256, K=256): 32 tiles, 1/wave.
// Candidate GEMM (32x128): 16 tiles on waves 0..15. State in LDS.
// Packed weights: Pg [8][256][32], Pc [8][128][32].
// ---------------------------------------------------------------------------
__global__ void __launch_bounds__(1024)
gru_seq_kernel(const float* __restrict__ xs, const int* __restrict__ len,
               const _Float16* __restrict__ Pg, const float* __restrict__ bg,
               const _Float16* __restrict__ Pc, const float* __restrict__ bc,
               float* __restrict__ ys, int Tlen, int ldy, int ycol0, int reverse) {
  __shared__ __align__(16) float h_lds[32 * 128];
  __shared__ __align__(16) float g_lds[32 * 256];
  __shared__ __align__(16) float c_lds[32 * 128];
  const int tid = threadIdx.x;
  const int w = tid >> 5, lane = tid & 31;
  for (int e = tid; e < 32 * 128; e += 1024) h_lds[e] = 0.f;
  __syncthreads();
  for (int s = 0; s < Tlen; ++s) {
    int t = reverse ? (Tlen - 1 - s) : s;
    {   // ---- gates: g = sigmoid([x,h] @ Wg + bg), 32x256 ----
      int m0 = (w >> 4) * 16;
      int n0 = (w & 15) * 16;
      int mb = m0 + (lane & 15);
      const float* xrow = xs + ((long)mb * Tlen + t) * 128;
      const float* hrow = &h_lds[mb * 128];
      int kh = (lane >> 4) * 8, kb = (lane >> 4) * 8;
      int col = n0 + (lane & 15);
      const _Float16* pb = Pg + (long)col * 32 + kb;
      v8f acc = {};
      for (int k0 = 0; k0 < 256; k0 += 32, pb += 256 * 32) {
        v4f x0, x1, x2, x3;
        if (k0 < 128) {
          int r0 = k0 + kh, r1 = k0 + 16 + kh;
          x0 = *(const v4f*)(xrow + r0); x1 = *(const v4f*)(xrow + r0 + 4);
          x2 = *(const v4f*)(xrow + r1); x3 = *(const v4f*)(xrow + r1 + 4);
        } else {
          int r0 = k0 - 128 + kh, r1 = k0 - 128 + 16 + kh;
          x0 = *(const v4f*)(hrow + r0); x1 = *(const v4f*)(hrow + r0 + 4);
          x2 = *(const v4f*)(hrow + r1); x3 = *(const v4f*)(hrow + r1 + 4);
        }
        v16h a; pack16(a, x0, x1, x2, x3);
        v16h bf = load_ph(pb);
        acc = WMMA16(acc, a, bf);
      }
      float bv = bg[col];
      int rb = (lane >> 4) * 8;
#pragma unroll
      for (int r = 0; r < 8; ++r)
        g_lds[(m0 + rb + r) * 256 + col] = 1.f / (1.f + expf(-(acc[r] + bv)));
    }
    __syncthreads();
    if (w < 16) {  // ---- candidate: c = tanh([x, r*h] @ Wc + bc), 32x128 ----
      int m0 = (w >> 3) * 16;
      int n0 = (w & 7) * 16;
      int mb = m0 + (lane & 15);
      const float* xrow = xs + ((long)mb * Tlen + t) * 128;
      const float* hrow = &h_lds[mb * 128];
      const float* grow = &g_lds[mb * 256];
      int kh = (lane >> 4) * 8, kb = (lane >> 4) * 8;
      int col = n0 + (lane & 15);
      const _Float16* pb = Pc + (long)col * 32 + kb;
      v8f acc = {};
      for (int k0 = 0; k0 < 256; k0 += 32, pb += 128 * 32) {
        v4f x0, x1, x2, x3;
        if (k0 < 128) {
          int r0 = k0 + kh, r1 = k0 + 16 + kh;
          x0 = *(const v4f*)(xrow + r0); x1 = *(const v4f*)(xrow + r0 + 4);
          x2 = *(const v4f*)(xrow + r1); x3 = *(const v4f*)(xrow + r1 + 4);
        } else {
          int r0 = k0 - 128 + kh, r1 = k0 - 128 + 16 + kh;
          x0 = *(const v4f*)(grow + r0) * *(const v4f*)(hrow + r0);
          x1 = *(const v4f*)(grow + r0 + 4) * *(const v4f*)(hrow + r0 + 4);
          x2 = *(const v4f*)(grow + r1) * *(const v4f*)(hrow + r1);
          x3 = *(const v4f*)(grow + r1 + 4) * *(const v4f*)(hrow + r1 + 4);
        }
        v16h a; pack16(a, x0, x1, x2, x3);
        v16h bf = load_ph(pb);
        acc = WMMA16(acc, a, bf);
      }
      float bv = bc[col];
      int rb = (lane >> 4) * 8;
#pragma unroll
      for (int r = 0; r < 8; ++r)
        c_lds[(m0 + rb + r) * 128 + col] = tanhf(acc[r] + bv);
    }
    __syncthreads();
    for (int e = tid; e < 32 * 128; e += 1024) {  // ---- update + emit ----
      int b = e >> 7, j = e & 127;
      float mm = len ? ((t < len[b]) ? 1.f : 0.f) : 1.f;
      float z = g_lds[b * 256 + 128 + j];
      float hv = h_lds[e];
      float hn = z * hv + (1.f - z) * c_lds[e];
      float nh = mm > 0.f ? hn : hv;
      h_lds[e] = nh;
      ys[((long)b * Tlen + t) * ldy + ycol0 + j] = nh * mm;
    }
    __syncthreads();
  }
}

// ---------------------------------------------------------------------------
// Attention: score -> masked softmax -> alpha (written straight to d_out).
// grid = B, block = T (128). LDS tree reductions.
// ---------------------------------------------------------------------------
__global__ void attention_kernel(const float* __restrict__ keys, const float* __restrict__ q,
                                 const float* __restrict__ v, const int* __restrict__ len,
                                 float* __restrict__ alpha_out, int step) {
  __shared__ float red[128];
  int b = blockIdx.x, t = threadIdx.x;
  const float* krow = keys + ((long)b * 128 + t) * 256;
  const float* qrow = q + b * 256;
  float s = 0.f;
  for (int d = 0; d < 256; d += 4) {
    v4f kv = *(const v4f*)(krow + d);
    v4f qv = *(const v4f*)(qrow + d);
    v4f vv = *(const v4f*)(v + d);
    s += tanhf(kv[0] + qv[0]) * vv[0] + tanhf(kv[1] + qv[1]) * vv[1] +
         tanhf(kv[2] + qv[2]) * vv[2] + tanhf(kv[3] + qv[3]) * vv[3];
  }
  float sc = (t < len[b]) ? s : -1e9f;
  red[t] = sc; __syncthreads();
  for (int o = 64; o > 0; o >>= 1) { if (t < o) red[t] = fmaxf(red[t], red[t + o]); __syncthreads(); }
  float mx = red[0]; __syncthreads();
  float e = expf(sc - mx);
  red[t] = e; __syncthreads();
  for (int o = 64; o > 0; o >>= 1) { if (t < o) red[t] += red[t + o]; __syncthreads(); }
  alpha_out[((long)b * 120 + step) * 128 + t] = e / red[0];
}

// ctx[b,d] = sum_t alpha[b,t] * memory[b,t,d]   (grid=B, block=256)
__global__ void ctx_kernel(const float* __restrict__ alpha, const float* __restrict__ memory,
                           float* __restrict__ ctx, int step) {
  int b = blockIdx.x, d = threadIdx.x;
  const float* al = alpha + ((long)b * 120 + step) * 128;
  float s = 0.f;
  for (int t = 0; t < 128; ++t) s += al[t] * memory[((long)b * 128 + t) * 256 + d];
  ctx[b * 256 + d] = s;
}

// ---- small glue kernels --------------------------------------------------
__global__ void gather_rows_kernel(const float* __restrict__ tab, const int* __restrict__ idx,
                                   float* __restrict__ out, int d) {
  out[(long)blockIdx.x * d + threadIdx.x] = tab[(long)idx[blockIdx.x] * d + threadIdx.x];
}
__global__ void fill_zero_kernel(float* p, int n) {
  int i = blockIdx.x * blockDim.x + threadIdx.x;
  if (i < n) p[i] = 0.f;
}
__global__ void concat3_kernel(const float* a, int da, const float* b, int db,
                               const float* c, int dc, float* dst) {
  int dt = da + db + dc;
  int idx = blockIdx.x * blockDim.x + threadIdx.x;
  if (idx >= 32 * dt) return;
  int row = idx / dt, col = idx - row * dt;
  float v;
  if (col < da) v = a[row * da + col];
  else if (col < da + db) v = b[row * db + (col - da)];
  else v = c[row * dc + (col - da - db)];
  dst[idx] = v;
}
__global__ void gru_xc_kernel(const float* x, int dx, const float* g, const float* h,
                              int dh, float* dst) {
  int dt = dx + dh;
  int idx = blockIdx.x * blockDim.x + threadIdx.x;
  if (idx >= 32 * dt) return;
  int row = idx / dt, col = idx - row * dt;
  dst[idx] = (col < dx) ? x[row * dx + col]
                        : g[row * 2 * dh + (col - dx)] * h[row * dh + (col - dx)];
}
__global__ void gru_combine_kernel(float* h, const float* g, const float* c, int dh) {
  int idx = blockIdx.x * blockDim.x + threadIdx.x;
  if (idx >= 32 * dh) return;
  int row = idx / dh, j = idx - row * dh;
  float z = g[row * 2 * dh + dh + j];
  h[idx] = z * h[idx] + (1.f - z) * c[idx];
}
__global__ void add_kernel(float* out, const float* a, const float* b, int n) {
  int i = blockIdx.x * blockDim.x + threadIdx.x;
  if (i < n) out[i] = a[i] + b[i];
}
__global__ void hw_combine_kernel(float* res, const float* hh, const float* tt, int n) {
  int i = blockIdx.x * blockDim.x + threadIdx.x;
  if (i < n) { float t = tt[i]; res[i] = hh[i] * t + res[i] * (1.f - t); }
}
__global__ void slice_ind_kernel(const float* frames, float* ind, int step) {
  int idx = blockIdx.x * blockDim.x + threadIdx.x;
  if (idx >= 32 * 80) return;
  int b = idx / 80, m = idx - b * 80;
  ind[idx] = frames[((long)step * 32 + b) * 400 + 320 + m];
}
__global__ void mel_reshape_kernel(const float* frames, float* mel) {
  int idx = blockIdx.x * blockDim.x + threadIdx.x;
  if (idx >= 32 * 600 * 80) return;
  int b = idx / (600 * 80);
  int rem = idx - b * 600 * 80;
  int t = rem / 80, m = rem - t * 80;
  int step = t / 5, r = t - step * 5;
  mel[idx] = frames[((long)step * 32 + b) * 400 + r * 80 + m];
}

// ---------------------------------------------------------------------------
// Host orchestration
// ---------------------------------------------------------------------------
extern "C" void kernel_launch(void* const* d_in, const int* in_sizes, int n_in,
                              void* d_out, int out_size, void* d_ws, size_t ws_size,
                              hipStream_t stream) {
  (void)in_sizes; (void)n_in; (void)out_size; (void)ws_size;
  auto f  = [&](int i) { return (const float*)d_in[i]; };
  auto ip = [&](int i) { return (const int*)d_in[i]; };

  float* outp = (float*)d_out;
  float* mel_out   = outp;                       // 32*600*80
  float* spec_out  = outp + 1536000;             // 32*600*513
  float* alpha_out = outp + 11385600;            // 32*120*128

  float* W = (float*)d_ws;
  size_t off = 0;
  auto alloc = [&](size_t n) { float* p = W + off; off += n; return p; };
  float* ws_emb    = alloc(4096 * 256);
  float* ws_pre1   = alloc(4096 * 256);
  float* ws_h      = alloc(4096 * 128);
  float* ws_stack  = alloc((size_t)19200 * 1024);   // also holds enc 4096x2048
  float* ws_proj1  = alloc((size_t)19200 * 256);
  float* ws_res    = alloc((size_t)19200 * 128);
  float* ws_res80  = alloc((size_t)19200 * 80);
  float* ws_hh     = alloc((size_t)19200 * 128);
  float* ws_tt     = alloc((size_t)19200 * 128);
  float* ws_mem    = alloc(4096 * 256);
  float* ws_keys   = alloc(4096 * 256);
  float* ws_post   = alloc((size_t)19200 * 256);
  float* ws_spk    = alloc(32 * 32);
  float* ws_frames = alloc((size_t)120 * 32 * 400);
  float* ws_ind    = alloc(32 * 80);
  float* ws_pp1    = alloc(32 * 256);
  float* ws_pp2    = alloc(32 * 128);
  float* ws_xatt   = alloc(32 * 160);
  float* ws_xcat   = alloc(32 * 544);
  float* ws_g      = alloc(32 * 512);
  float* ws_c      = alloc(32 * 256);
  float* ws_hatt   = alloc(32 * 256);
  float* ws_h1     = alloc(32 * 256);
  float* ws_h2     = alloc(32 * 256);
  float* ws_q      = alloc(32 * 256);
  float* ws_ctx    = alloc(32 * 256);
  float* ws_acoin  = alloc(32 * 256);
  float* ws_o1     = alloc(32 * 256);
  float* ws_o2     = alloc(32 * 256);

  // f16 packed-weight arena after the f32 region
  _Float16* H = (_Float16*)(W + off);
  size_t hoff = 0;
  auto halloc = [&](size_t n) { _Float16* p = H + hoff; hoff += (n + 15) & ~(size_t)15; return p; };
  auto ew = [&](size_t n) { return dim3((unsigned)((n + 255) / 256)); };
  auto packw = [&](const float* w, int width, int K, int N, int Npad) {
    int nch = (K + 31) / 32;
    size_t per = (size_t)nch * Npad * 32;
    size_t tot = per * width;
    _Float16* p = halloc(tot);
    pack_w_kernel<<<ew(tot), 256, 0, stream>>>(w, p, width, K, N, Npad, nch);
    return p;
  };

  // ---- pack every weight once per call ----
  _Float16* p_pre1   = packw(f(1), 1, 256, 256, 256);
  _Float16* p_pre2   = packw(f(3), 1, 256, 128, 128);
  _Float16* p_bank[16];
  for (int k = 1; k <= 16; ++k) p_bank[k - 1] = packw(f(5 + 2 * (k - 1)), k, 128, 128, 128);
  _Float16* p_eproj1 = packw(f(37), 3, 2048, 128, 128);
  _Float16* p_eproj2 = packw(f(39), 3, 128, 128, 128);
  _Float16* p_ehw[8];
  for (int l = 0; l < 4; ++l) {
    p_ehw[2 * l]     = packw(f(41 + 4 * l), 1, 128, 128, 128);
    p_ehw[2 * l + 1] = packw(f(43 + 4 * l), 1, 128, 128, 128);
  }
  _Float16* p_egf_g = packw(f(57), 1, 256, 256, 256);
  _Float16* p_egf_c = packw(f(59), 1, 256, 128, 128);
  _Float16* p_egb_g = packw(f(61), 1, 256, 256, 256);
  _Float16* p_egb_c = packw(f(63), 1, 256, 128, 128);
  _Float16* p_Wq    = packw(f(65), 1, 256, 256, 256);
  _Float16* p_Wm    = packw(f(66), 1, 256, 256, 256);
  _Float16* p_dW1   = packw(f(70), 1, 80, 256, 256);
  _Float16* p_dW2   = packw(f(72), 1, 256, 128, 128);
  _Float16* p_attg  = packw(f(74), 1, 416, 512, 512);
  _Float16* p_attc  = packw(f(76), 1, 416, 256, 256);
  _Float16* p_Wa    = packw(f(78), 1, 544, 256, 256);
  _Float16* p_g1g   = packw(f(80), 1, 512, 512, 512);
  _Float16* p_g1c   = packw(f(82), 1, 512, 256, 256);
  _Float16* p_g2g   = packw(f(84), 1, 512, 512, 512);
  _Float16* p_g2c   = packw(f(86), 1, 512, 256, 256);
  _Float16* p_mel   = packw(f(88), 1, 256, 400, 400);
  _Float16* p_pbank[8];
  for (int k = 1; k <= 8; ++k) p_pbank[k - 1] = packw(f(90 + 2 * (k - 1)), k, 80, 128, 128);
  _Float16* p_pproj1 = packw(f(106), 3, 1024, 256, 256);
  _Float16* p_pproj2 = packw(f(108), 3, 256, 80, 80);
  _Float16* p_hwin   = packw(f(110), 1, 80, 128, 128);
  _Float16* p_phw[8];
  for (int l = 0; l < 4; ++l) {
    p_phw[2 * l]     = packw(f(112 + 4 * l), 1, 128, 128, 128);
    p_phw[2 * l + 1] = packw(f(114 + 4 * l), 1, 128, 128, 128);
  }
  _Float16* p_pgf_g = packw(f(128), 1, 256, 256, 256);
  _Float16* p_pgf_c = packw(f(130), 1, 256, 128, 128);
  _Float16* p_pgb_g = packw(f(132), 1, 256, 256, 256);
  _Float16* p_pgb_c = packw(f(134), 1, 256, 128, 128);
  _Float16* p_spec  = packw(f(136), 1, 256, 513, 528);

  auto gemm = [&](const float* A, const _Float16* P, const float* bias, const float* resid,
                  float* C, int M, int N, int K, int ldc, int act) {
    int Npad = (N + 15) & ~15;
    dim3 g((Npad + 31) / 32, M / 16);
    gemm_kernel<<<g, 32, 0, stream>>>(A, P, bias, resid, C, N, Npad, K, ldc, act);
  };
  auto conv = [&](const float* X, const _Float16* P, const float* bias, const float* resid,
                  float* Y, int Tlen, int Cin, int Cout, int width, int pad,
                  int ldy, int ycol0, int act, int pool) {
    dim3 g((Cout + 31) / 32, (32 * Tlen) / 16);
    conv_gemm_kernel<<<g, 32, 0, stream>>>(X, P, bias, resid, Y, Tlen, Cin, Cout,
                                           width, pad, ldy, ycol0, act, pool);
  };
  auto highway = [&](float* res, int M, const _Float16* Ph, const float* bh,
                     const _Float16* Pt, const float* bt) {
    gemm(res, Ph, bh, nullptr, ws_hh, M, 128, 128, 128, ACT_RELU);
    gemm(res, Pt, bt, nullptr, ws_tt, M, 128, 128, 128, ACT_SIG);
    size_t n = (size_t)M * 128;
    hw_combine_kernel<<<ew(n), 256, 0, stream>>>(res, ws_hh, ws_tt, (int)n);
  };
  auto gru_step = [&](float* h, const float* x, int dx, int dh,
                      const _Float16* Pg, const float* bg, const _Float16* Pc, const float* bc) {
    int dt = dx + dh;
    concat3_kernel<<<ew(32 * dt), 256, 0, stream>>>(x, dx, h, dh, nullptr, 0, ws_xcat);
    gemm(ws_xcat, Pg, bg, nullptr, ws_g, 32, 2 * dh, dt, 2 * dh, ACT_SIG);
    gru_xc_kernel<<<ew(32 * dt), 256, 0, stream>>>(x, dx, ws_g, h, dh, ws_xcat);
    gemm(ws_xcat, Pc, bc, nullptr, ws_c, 32, dh, dt, dh, ACT_TANH);
    gru_combine_kernel<<<ew(32 * dh), 256, 0, stream>>>(h, ws_g, ws_c, dh);
  };

  // ------------------ Encoder ------------------
  gather_rows_kernel<<<4096, 256, 0, stream>>>(f(0), ip(138), ws_emb, 256);
  gemm(ws_emb, p_pre1, f(2), nullptr, ws_pre1, 4096, 256, 256, 256, ACT_RELU);
  gemm(ws_pre1, p_pre2, f(4), nullptr, ws_h, 4096, 128, 256, 128, ACT_RELU);
  for (int k = 1; k <= 16; ++k)
    conv(ws_h, p_bank[k - 1], f(6 + 2 * (k - 1)), nullptr, ws_stack,
         128, 128, 128, k, (k - 1) / 2, 2048, (k - 1) * 128, ACT_RELU, 0);
  conv(ws_stack, p_eproj1, f(38), nullptr, ws_proj1, 128, 2048, 128, 3, 1, 128, 0, ACT_RELU, 1);
  conv(ws_proj1, p_eproj2, f(40), ws_h, ws_res, 128, 128, 128, 3, 1, 128, 0, ACT_NONE, 0);
  for (int l = 0; l < 4; ++l)
    highway(ws_res, 4096, p_ehw[2 * l], f(42 + 4 * l), p_ehw[2 * l + 1], f(44 + 4 * l));
  gru_seq_kernel<<<1, 1024, 0, stream>>>(ws_res, ip(139), p_egf_g, f(58), p_egf_c, f(60),
                                         ws_mem, 128, 256, 0, 0);
  gru_seq_kernel<<<1, 1024, 0, stream>>>(ws_res, ip(139), p_egb_g, f(62), p_egb_c, f(64),
                                         ws_mem, 128, 256, 128, 1);
  gemm(ws_mem, p_Wm, nullptr, nullptr, ws_keys, 4096, 256, 256, 256, ACT_NONE);
  gather_rows_kernel<<<32, 32, 0, stream>>>(f(69), ip(140), ws_spk, 32);

  // ------------------ Decoder (120 sequential steps) ------------------
  fill_zero_kernel<<<ew(32 * 256), 256, 0, stream>>>(ws_hatt, 32 * 256);
  fill_zero_kernel<<<ew(32 * 256), 256, 0, stream>>>(ws_h1, 32 * 256);
  fill_zero_kernel<<<ew(32 * 256), 256, 0, stream>>>(ws_h2, 32 * 256);
  fill_zero_kernel<<<ew(32 * 80), 256, 0, stream>>>(ws_ind, 32 * 80);
  for (int s = 0; s < 120; ++s) {
    gemm(ws_ind, p_dW1, f(71), nullptr, ws_pp1, 32, 256, 80, 256, ACT_RELU);
    gemm(ws_pp1, p_dW2, f(73), nullptr, ws_pp2, 32, 128, 256, 128, ACT_RELU);
    concat3_kernel<<<ew(32 * 160), 256, 0, stream>>>(ws_pp2, 128, ws_spk, 32, nullptr, 0, ws_xatt);
    gru_step(ws_hatt, ws_xatt, 160, 256, p_attg, f(75), p_attc, f(77));
    gemm(ws_hatt, p_Wq, f(67), nullptr, ws_q, 32, 256, 256, 256, ACT_NONE);
    attention_kernel<<<32, 128, 0, stream>>>(ws_keys, ws_q, f(68), ip(139), alpha_out, s);
    ctx_kernel<<<32, 256, 0, stream>>>(alpha_out, ws_mem, ws_ctx, s);
    concat3_kernel<<<ew(32 * 544), 256, 0, stream>>>(ws_hatt, 256, ws_spk, 32, ws_ctx, 256, ws_xcat);
    gemm(ws_xcat, p_Wa, f(79), nullptr, ws_acoin, 32, 256, 544, 256, ACT_NONE);
    gru_step(ws_h1, ws_acoin, 256, 256, p_g1g, f(81), p_g1c, f(83));
    add_kernel<<<ew(32 * 256), 256, 0, stream>>>(ws_o1, ws_h1, ws_acoin, 32 * 256);
    gru_step(ws_h2, ws_o1, 256, 256, p_g2g, f(85), p_g2c, f(87));
    add_kernel<<<ew(32 * 256), 256, 0, stream>>>(ws_o2, ws_h2, ws_o1, 32 * 256);
    gemm(ws_o2, p_mel, f(89), nullptr, ws_frames + (size_t)s * 32 * 400, 32, 400, 256, 400, ACT_NONE);
    slice_ind_kernel<<<ew(32 * 80), 256, 0, stream>>>(ws_frames, ws_ind, s);
  }
  mel_reshape_kernel<<<ew(32 * 600 * 80), 256, 0, stream>>>(ws_frames, mel_out);

  // ------------------ Postnet CBHG ------------------
  const float* melp = mel_out;
  for (int k = 1; k <= 8; ++k)
    conv(melp, p_pbank[k - 1], f(91 + 2 * (k - 1)), nullptr, ws_stack,
         600, 80, 128, k, (k - 1) / 2, 1024, (k - 1) * 128, ACT_RELU, 0);
  conv(ws_stack, p_pproj1, f(107), nullptr, ws_proj1, 600, 1024, 256, 3, 1, 256, 0, ACT_RELU, 1);
  conv(ws_proj1, p_pproj2, f(109), melp, ws_res80, 600, 256, 80, 3, 1, 80, 0, ACT_NONE, 0);
  gemm(ws_res80, p_hwin, f(111), nullptr, ws_res, 19200, 128, 80, 128, ACT_NONE);
  for (int l = 0; l < 4; ++l)
    highway(ws_res, 19200, p_phw[2 * l], f(113 + 4 * l), p_phw[2 * l + 1], f(115 + 4 * l));
  gru_seq_kernel<<<1, 1024, 0, stream>>>(ws_res, nullptr, p_pgf_g, f(129), p_pgf_c, f(131),
                                         ws_post, 600, 256, 0, 0);
  gru_seq_kernel<<<1, 1024, 0, stream>>>(ws_res, nullptr, p_pgb_g, f(133), p_pgb_c, f(135),
                                         ws_post, 600, 256, 128, 1);
  gemm(ws_post, p_spec, f(137), nullptr, spec_out, 19200, 513, 256, 513, ACT_NONE);
}
